// SentAttention_6176162972190
// MI455X (gfx1250) — matrix operand order repaired and checked by reading, not compiled, not run on hardware
//
#include <hip/hip_runtime.h>
#include <hip/hip_bf16.h>

#define H   10
#define D   20          // 2*H
#define G3  30          // 3*H
#define CLS 90
#define NB  4096        // batch
#define NS  512         // steps

typedef __attribute__((ext_vector_type(16))) _Float16 v16h;
typedef __attribute__((ext_vector_type(8)))  _Float16 v8h;
typedef __attribute__((ext_vector_type(8)))  float    v8f;

union V16 { v16h v; v8h h[2]; };

// ---------------------------------------------------------------------------
// Kernel 1: fused bi-GRU scan + online-softmax attention.
//   - 64 threads (2 waves). 32 batch rows per block (wave w owns a 16-row
//     WMMA M-tile). Thread t owns (row = t>>1, direction = t&1) for the
//     recurrence; lane pairs exchange via __shfl_xor.
//   - Per step: stage x[32x20] -> LDS f16 (rows padded to 32 halves so the
//     WMMA A-fragment is two ds_load_b128 per lane), 4x v_wmma_f32_16x16x32_f16
//     produce all 60 gate pre-activations, scatter to LDS, then VALU GRU cell
//     + online softmax accumulation.
// ---------------------------------------------------------------------------
#define K1_ROWS 32
#define K1_THR  64

__global__ __launch_bounds__(K1_THR)
void gru_att_kernel(const float* __restrict__ s,
                    const float* __restrict__ Wih_f, const float* __restrict__ Whh_f,
                    const float* __restrict__ bih_f, const float* __restrict__ bhh_f,
                    const float* __restrict__ Wih_b, const float* __restrict__ Whh_b,
                    const float* __restrict__ bih_b, const float* __restrict__ bhh_b,
                    const float* __restrict__ W_att, const float* __restrict__ b_att,
                    const float* __restrict__ W_ctx,
                    float* __restrict__ sent)
{
    __shared__ _Float16 lds_x[K1_ROWS * 32];      // rows padded to 32 halves (K-pad = 0)
    __shared__ float    lds_gx[2 * 16 * 64];      // per-wave 16x64 gate tile
    __shared__ float    lds_w[1100];              // Whh(600) bhh(60) W_att(400) b_att(20) wctx(20)

    const int tid   = threadIdx.x;
    const int wave  = tid >> 5;
    const int lane  = tid & 31;
    const int ln    = lane & 15;
    const int khalf = lane >> 4;
    const int rowbase = blockIdx.x * K1_ROWS;

    // ---- cooperative weight load into LDS ----
    for (int i = tid; i < 300; i += K1_THR) { lds_w[i] = Whh_f[i]; lds_w[300 + i] = Whh_b[i]; }
    for (int i = tid; i < 30;  i += K1_THR) { lds_w[600 + i] = bhh_f[i]; lds_w[630 + i] = bhh_b[i]; }
    for (int i = tid; i < 400; i += K1_THR) { lds_w[660 + i] = W_att[i]; }
    for (int i = tid; i < 20;  i += K1_THR) { lds_w[1060 + i] = b_att[i]; lds_w[1080 + i] = W_ctx[i]; }
    for (int i = tid; i < K1_ROWS * 32; i += K1_THR) lds_x[i] = (_Float16)0.f;   // K padding stays 0

    // ---- build B fragments for Wih (both directions concatenated: 60 cols) ----
    // B 32x16 layout: element e -> K = (e&7) + 16*(e>>3) + 8*khalf, N = tile*16 + ln
    V16   bfrag[4];
    float biasv[4];
    #pragma unroll
    for (int t = 0; t < 4; ++t) {
        const int n = t * 16 + ln;
        #pragma unroll
        for (int e = 0; e < 16; ++e) {
            const int k = (e & 7) + 16 * (e >> 3) + 8 * khalf;
            float w = 0.f;
            if (k < D && n < 2 * G3)
                w = (n < G3) ? Wih_f[n * D + k] : Wih_b[(n - G3) * D + k];
            bfrag[t].v[e] = (_Float16)w;
        }
        float bv = 0.f;
        if (n < G3) bv = bih_f[n]; else if (n < 2 * G3) bv = bih_b[n - G3];
        biasv[t] = bv;
    }

    // recurrence ownership
    const int rrow = tid >> 1;          // local row 0..31
    const int dir  = tid & 1;           // 0 = forward cell, 1 = backward cell
    const int wv   = rrow >> 4, mloc = rrow & 15;
    const float* whh = &lds_w[dir * 300];
    const float* bhh = &lds_w[600 + dir * 30];

    float h[H], acc[H];
    #pragma unroll
    for (int i = 0; i < H; ++i) { h[i] = 0.f; acc[i] = 0.f; }
    float mrun = -3.4e38f, lrun = 0.f;

    __syncthreads();

    const float* srcbase = s + ((size_t)(rowbase + rrow) * NS) * D + dir * H;

    for (int step = 0; step < NS; ++step) {
        // ---- stage x (this step, my row, my half) into LDS as f16 ----
        {
            const float* src = srcbase + (size_t)step * D;
            _Float16* dst = &lds_x[rrow * 32 + dir * H];
            #pragma unroll
            for (int i = 0; i < H; ++i) dst[i] = (_Float16)src[i];
            if (step + 8 < NS) __builtin_prefetch(src + 8 * D, 0, 1);
        }
        __syncthreads();

        // ---- WMMA: gx = x @ [Wih_f;Wih_b]^T  (one K=32 WMMA per 16-col tile) ----
        {
            const int arow = wave * 16 + ln;
            V16 a;
            a.h[0] = *(const v8h*)&lds_x[arow * 32 + khalf * 8];        // K = 8*khalf..+7
            a.h[1] = *(const v8h*)&lds_x[arow * 32 + 16 + khalf * 8];   // K = 16+8*khalf..+7
            #pragma unroll
            for (int t = 0; t < 4; ++t) {
                v8f c = {};
                c = __builtin_amdgcn_wmma_f32_16x16x32_f16(false, a.v, false, bfrag[t].v,
                                                           (short)0, c, false, false);
                float* g = &lds_gx[wave * 1024 + t * 16 + ln];
                #pragma unroll
                for (int r = 0; r < 8; ++r)
                    g[(r + 8 * khalf) * 64] = c[r] + biasv[t];          // M = r + 8*khalf
            }
        }
        __syncthreads();

        // ---- GRU cell + online-softmax attention for my (row, direction) ----
        {
            const float* gx = &lds_gx[wv * 1024 + mloc * 64 + dir * G3];
            float gh[G3];
            #pragma unroll
            for (int j = 0; j < G3; ++j) {
                float a0 = bhh[j];
                #pragma unroll
                for (int k = 0; k < H; ++k) a0 += whh[j * H + k] * h[k];
                gh[j] = a0;
            }
            #pragma unroll
            for (int i = 0; i < H; ++i) {
                float r  = 1.f / (1.f + __expf(-(gx[i] + gh[i])));
                float z  = 1.f / (1.f + __expf(-(gx[H + i] + gh[H + i])));
                float nn = gx[2 * H + i] + r * gh[2 * H + i];
                nn = 1.f - 2.f / (__expf(2.f * nn) + 1.f);              // tanh
                h[i] = (1.f - z) * nn + z * h[i];
            }
            // u_j = tanh(sum_k W_att[j,k]*out_k + b_att[j]); out split across lane pair
            float p[2 * H];
            #pragma unroll
            for (int j = 0; j < 2 * H; ++j) {
                float a0 = 0.f;
                const float* wr = &lds_w[660 + j * (2 * H) + dir * H];
                #pragma unroll
                for (int k = 0; k < H; ++k) a0 += wr[k] * h[k];
                p[j] = a0 + __shfl_xor(a0, 1);                          // full 20-wide dot
            }
            float lpart = 0.f;
            #pragma unroll
            for (int i = 0; i < H; ++i) {
                float pv = dir ? p[H + i] : p[i];                       // my j-half
                float u  = pv + lds_w[1060 + dir * H + i];
                u = 1.f - 2.f / (__expf(2.f * u) + 1.f);
                lpart += lds_w[1080 + dir * H + i] * u;
            }
            const float logit = lpart + __shfl_xor(lpart, 1);
            // online softmax update (identical in both lanes of the pair)
            const float nm = fmaxf(mrun, logit);
            const float sc = __expf(mrun - nm);
            const float pe = __expf(logit - nm);
            lrun = lrun * sc + pe;
            #pragma unroll
            for (int i = 0; i < H; ++i) acc[i] = acc[i] * sc + pe * h[i];
            mrun = nm;
        }
        __syncthreads();
    }

    const float inv = 1.f / lrun;
    float* o = &sent[(size_t)(rowbase + rrow) * D + dir * H];
    #pragma unroll
    for (int i = 0; i < H; ++i) o[i] = acc[i] * inv;
}

// ---------------------------------------------------------------------------
// Kernel 2: classifier  out = relu(sent@W_c1^T+b1) @ W_c2^T + b2
//   One wave per 16-row tile. GEMM1: K=32 (pad of 20), N=96 (pad of 90) ->
//   6 WMMA; relu -> LDS f16; GEMM2: K=96 = 3 chunks x 6 N-tiles -> 18 WMMA.
// ---------------------------------------------------------------------------
__global__ __launch_bounds__(32)
void cls_kernel(const float* __restrict__ sent,
                const float* __restrict__ W_c1, const float* __restrict__ b_c1,
                const float* __restrict__ W_c2, const float* __restrict__ b_c2,
                float* __restrict__ out)
{
    __shared__ _Float16 lds_h[16 * 96];
    const int lane = threadIdx.x & 31;
    const int ln = lane & 15, khalf = lane >> 4;
    const int rowbase = blockIdx.x * 16;

    // A1 fragment from sent (A 16x32 layout: K = (e&7) + 16*(e>>3) + 8*khalf)
    V16 a1;
    #pragma unroll
    for (int e = 0; e < 16; ++e) {
        const int k = (e & 7) + 16 * (e >> 3) + 8 * khalf;
        a1.v[e] = (k < D) ? (_Float16)sent[(size_t)(rowbase + ln) * D + k] : (_Float16)0.f;
    }

    // hidden = relu(sent @ W_c1^T + b_c1), stored to LDS as f16 [16][96]
    #pragma unroll
    for (int t = 0; t < 6; ++t) {
        const int n = t * 16 + ln;
        V16 b1;
        #pragma unroll
        for (int e = 0; e < 16; ++e) {
            const int k = (e & 7) + 16 * (e >> 3) + 8 * khalf;
            b1.v[e] = (k < D && n < CLS) ? (_Float16)W_c1[n * D + k] : (_Float16)0.f;
        }
        v8f c = {};
        c = __builtin_amdgcn_wmma_f32_16x16x32_f16(false, a1.v, false, b1.v,
                                                   (short)0, c, false, false);
        const float bias = (n < CLS) ? b_c1[n] : 0.f;
        #pragma unroll
        for (int r = 0; r < 8; ++r) {
            const float hv = fmaxf(c[r] + bias, 0.f);                   // pad cols -> 0
            lds_h[(r + 8 * khalf) * 96 + n] = (_Float16)hv;
        }
    }
    __syncthreads();

    // A2 fragments: 3 K-chunks of 32 over hidden[16][96] (rows 192B -> b128 loads)
    V16 a2[3];
    #pragma unroll
    for (int cc = 0; cc < 3; ++cc) {
        a2[cc].h[0] = *(const v8h*)&lds_h[ln * 96 + cc * 32 + khalf * 8];
        a2[cc].h[1] = *(const v8h*)&lds_h[ln * 96 + cc * 32 + 16 + khalf * 8];
    }

    #pragma unroll
    for (int t = 0; t < 6; ++t) {
        const int n = t * 16 + ln;
        v8f c = {};
        #pragma unroll
        for (int cc = 0; cc < 3; ++cc) {
            V16 b2;
            #pragma unroll
            for (int e = 0; e < 16; ++e) {
                const int k = cc * 32 + (e & 7) + 16 * (e >> 3) + 8 * khalf;
                b2.v[e] = (k < CLS && n < CLS) ? (_Float16)W_c2[n * CLS + k] : (_Float16)0.f;
            }
            c = __builtin_amdgcn_wmma_f32_16x16x32_f16(false, a2[cc].v, false, b2.v,
                                                       (short)0, c, false, false);
        }
        if (n < CLS) {
            const float bias = b_c2[n];
            #pragma unroll
            for (int r = 0; r < 8; ++r)
                out[(size_t)(rowbase + r + 8 * khalf) * CLS + n] = c[r] + bias;
        }
    }
}

// ---------------------------------------------------------------------------
extern "C" void kernel_launch(void* const* d_in, const int* in_sizes, int n_in,
                              void* d_out, int out_size, void* d_ws, size_t ws_size,
                              hipStream_t stream) {
    (void)in_sizes; (void)n_in; (void)out_size; (void)ws_size;
    const float* s     = (const float*)d_in[0];
    const float* Wih_f = (const float*)d_in[1];
    const float* Whh_f = (const float*)d_in[2];
    const float* bih_f = (const float*)d_in[3];
    const float* bhh_f = (const float*)d_in[4];
    const float* Wih_b = (const float*)d_in[5];
    const float* Whh_b = (const float*)d_in[6];
    const float* bih_b = (const float*)d_in[7];
    const float* bhh_b = (const float*)d_in[8];
    const float* W_att = (const float*)d_in[9];
    const float* b_att = (const float*)d_in[10];
    const float* W_ctx = (const float*)d_in[11];
    const float* W_c1  = (const float*)d_in[12];
    const float* b_c1  = (const float*)d_in[13];
    const float* W_c2  = (const float*)d_in[14];
    const float* b_c2  = (const float*)d_in[15];

    float* sent = (float*)d_ws;   // [4096, 20] f32 scratch (~328 KB)

    gru_att_kernel<<<NB / K1_ROWS, K1_THR, 0, stream>>>(
        s, Wih_f, Whh_f, bih_f, bhh_f, Wih_b, Whh_b, bih_b, bhh_b,
        W_att, b_att, W_ctx, sent);

    cls_kernel<<<NB / 16, 32, 0, stream>>>(sent, W_c1, b_c1, W_c2, b_c2, (float*)d_out);
}
